// nh_Block_layer_8813272891664
// MI455X (gfx1250) — matrix operand order repaired and checked by reading, not compiled, not run on hardware
//
#include <hip/hip_runtime.h>
#include <hip/hip_bf16.h>

// ---------------- problem constants ----------------
#define NH     16
#define MD     256
#define MDNH   64
#define FF     128
#define NHEADS 4
#define BB     2
#define TT     8192
#define TCC    8192
#define BT     (BB*TT)     // 16384
#define BTC    (BB*TCC)    // 16384

// LDS leading dims (halves), multiples of 8 so float4 (16B) access stays aligned
#define LDX 72    // 64-wide activation rows
#define LDY 136   // 128-wide ff1 rows

// packed-weight fragment offsets in halves (each frag = 32 lanes * 16 halves = 512)
#define PK_WRED   0         // 8 kt * 4 nt = 32 frags
#define PK_WOUT   16384     // 2 kt * 4 nt = 8 frags
#define PK_WFF1   20480     // 2 kt * 8 nt = 16 frags
#define PK_WFEAT  28672     // 32 kt * 16 nt = 512 frags
#define PK_CONV   290816    // 1 A-fragment (zero padded K=16->32)
#define PK_FRAGS  569

// workspace byte offsets
#define WS_HH   0                    // 2*8192*64 halves = 2 MB
#define WS_IDX  2097152              // 16384*16 ints   = 1 MB
#define WS_PK   3145728              // packed weights  ~ 0.57 MB

typedef __attribute__((ext_vector_type(16))) _Float16 v16h;
typedef __attribute__((ext_vector_type(8)))  float    v8f;

__device__ __forceinline__ float leaky(float v) { return v >= 0.f ? v : 0.2f * v; }

// ---- fragment loaders -------------------------------------------------------
// Packed B (or conv-A) fragment: 32 contiguous bytes per lane, fully coalesced.
__device__ __forceinline__ v16h load_pk32(const _Float16* pk, int frag, int lane) {
    return *((const v16h*)pk + (frag * 32 + lane));
}

// A 16x32 f16 fragment from a row-major LDS matrix (ISA layout:
// lanes 0-15: K 0-7 / 16-23 ; lanes 16-31: K 8-15 / 24-31 ; m = lane&15).
__device__ __forceinline__ v16h load_a_rowmajor(const _Float16* base, int ld, int lane) {
    const int m = lane & 15, hf = lane >> 4;
    const _Float16* p = base + m * ld + hf * 8;
    v16h a;
    *(float4*)&a       = *(const float4*)p;         // K group 0
    *((float4*)&a + 1) = *(const float4*)(p + 16);  // K group 1
    return a;
}

// ---------------- kernel: pack weights into WMMA fragment order --------------
__global__ void k_pack(const float* __restrict__ w_red, const float* __restrict__ w_out,
                       const float* __restrict__ w_ff1, const float* __restrict__ conv_w,
                       const float* __restrict__ w_feat, _Float16* __restrict__ pk) {
    const int fid  = blockIdx.x;
    const int lane = threadIdx.x & 31;
    const int n16  = lane & 15, hf = lane >> 4;

    if (fid == 568) {
        // conv A-fragment: A[m][k] = conv_w[m][k] for k<16 else 0  (K padded to 32)
        _Float16* d = pk + PK_CONV + lane * 16;
#pragma unroll
        for (int i = 0; i < 16; ++i) {
            int k = hf * 16 + i;
            d[i] = (_Float16)((k < 16) ? conv_w[n16 * 16 + k] : 0.f);
        }
        return;
    }

    const float* src; int Ktot, NT, local; _Float16* dst;
    if (fid < 32)       { src = w_red;  Ktot = MD;   NT = 4;  local = fid;      dst = pk + PK_WRED;  }
    else if (fid < 40)  { src = w_out;  Ktot = MDNH; NT = 4;  local = fid - 32; dst = pk + PK_WOUT;  }
    else if (fid < 56)  { src = w_ff1;  Ktot = MDNH; NT = 8;  local = fid - 40; dst = pk + PK_WFF1;  }
    else                { src = w_feat; Ktot = 1024; NT = 16; local = fid - 56; dst = pk + PK_WFEAT; }

    const int kt = local / NT, nt = local % NT;
    _Float16* d = dst + local * 512 + lane * 16;
#pragma unroll
    for (int i = 0; i < 16; ++i) {
        // B operand layout: lanes 0-15 hold K 0-7 / 16-23 of the tile, lanes 16-31 hold K 8-15 / 24-31.
        int k = kt * 32 + ((i < 8) ? (hf * 8 + i) : (16 + hf * 8 + (i - 8)));
        int n = nt * 16 + n16;
        d[i] = (_Float16)src[(size_t)n * Ktot + k];  // B[k][n] = W[n][k] (W row-major [N][K])
    }
}

// ---------------- kernel 1: h = LN1(leaky(x_cross @ w_red^T)) -> f16 ---------
__global__ void k_reduce_ln1(const float* __restrict__ xc, const _Float16* __restrict__ pk,
                             const float* __restrict__ g1, const float* __restrict__ b1,
                             _Float16* __restrict__ hh) {
    __attribute__((aligned(16))) __shared__ _Float16 xa[64 * 264];
    __attribute__((aligned(16))) __shared__ float    hc[64 * 72];
    const int tid = threadIdx.x;        // 128 threads, 4 waves
    const int r0  = blockIdx.x * 64;

    for (int e = tid; e < 64 * 256; e += 128) {
        int r = e >> 8, c = e & 255;
        xa[r * 264 + c] = (_Float16)xc[(size_t)(r0 + r) * MD + c];
    }
    __syncthreads();

    const int wave = tid >> 5, lane = tid & 31;
    const int n16 = lane & 15, hf = lane >> 4;
    v8f acc[4] = {};
#pragma unroll
    for (int kt = 0; kt < 8; ++kt) {
        v16h a = load_a_rowmajor(xa + (wave * 16) * 264 + kt * 32, 264, lane);
#pragma unroll
        for (int nt = 0; nt < 4; ++nt) {
            v16h bfr = load_pk32(pk + PK_WRED, kt * 4 + nt, lane);
            acc[nt] = __builtin_amdgcn_wmma_f32_16x16x32_f16(false, a, false, bfr,
                                                             (short)0, acc[nt], false, false);
        }
    }
#pragma unroll
    for (int nt = 0; nt < 4; ++nt)
#pragma unroll
        for (int i = 0; i < 8; ++i)
            hc[(wave * 16 + i + 8 * hf) * 72 + nt * 16 + n16] = leaky(acc[nt][i]);
    __syncthreads();

    if (tid < 64) {
        float s = 0.f, s2 = 0.f;
#pragma unroll
        for (int d = 0; d < 64; ++d) { float v = hc[tid * 72 + d]; s += v; s2 += v * v; }
        float m  = s * (1.f / 64.f);
        float rs = rsqrtf(s2 * (1.f / 64.f) - m * m + 1e-5f);
        _Float16* dst = hh + (size_t)(r0 + tid) * MDNH;
        for (int d = 0; d < 64; ++d)
            dst[d] = (_Float16)((hc[tid * 72 + d] - m) * rs * g1[d] + b1[d]);
    }
}

// ---------------- kernel 2: 16-NN search (streaming top-16 per thread) -------
__global__ void k_knn(const float* __restrict__ coords, const float* __restrict__ coords_cross,
                      int* __restrict__ idxb) {
    __shared__ float cc[2048 * 2];
    const int tid = threadIdx.x;
    const int q   = blockIdx.x * 256 + tid;    // T=8192 -> batch uniform per block
    const int b   = q / TT;
    const float qx = coords[(size_t)q * 2], qy = coords[(size_t)q * 2 + 1];

    float nd[16]; int ni[16];
#pragma unroll
    for (int j = 0; j < 16; ++j) { nd[j] = 3.0e38f; ni[j] = 0; }

    for (int ch = 0; ch < TCC; ch += 2048) {
        __syncthreads();
        for (int e = tid; e < 2048 * 2; e += 256)
            cc[e] = coords_cross[((size_t)b * TCC + ch) * 2 + e];
        __syncthreads();
        for (int c = 0; c < 2048; ++c) {
            float dx = cc[2 * c] - qx, dy = cc[2 * c + 1] - qy;
            float d  = dx * dx + dy * dy;
            if (d < nd[15]) {                      // rare once threshold tightens
                nd[15] = d; ni[15] = ch + c;
#pragma unroll
                for (int j = 15; j > 0; --j) {
                    if (nd[j] < nd[j - 1]) {
                        float td = nd[j]; nd[j] = nd[j - 1]; nd[j - 1] = td;
                        int   ti = ni[j]; ni[j] = ni[j - 1]; ni[j - 1] = ti;
                    }
                }
            }
        }
    }
#pragma unroll
    for (int j = 0; j < 16; ++j) idxb[(size_t)q * NH + j] = ni[j];
}

// ---------------- kernel 3: fused gather/MHA/out/ff/conv/feat/LN3 ------------
__global__ void k_fused(const _Float16* __restrict__ hh, const int* __restrict__ idxb,
                        const _Float16* __restrict__ pk, const float* __restrict__ logit_scale,
                        const float* __restrict__ g2, const float* __restrict__ b2,
                        const float* __restrict__ wff2,
                        const float* __restrict__ g3, const float* __restrict__ b3,
                        float* __restrict__ out) {
    extern __shared__ char smem[];
    _Float16* bufA = (_Float16*)smem;            // [256][LDX]  activations
    _Float16* bufB = bufA + 256 * LDX;           // [256][LDX]  att_out / conv out
    _Float16* yscr = bufB + 256 * LDX;           // [256][LDY]  ff1 output
    float*    wf2  = (float*)(yscr + 256 * LDY); // [128]
    float*    outb = wf2 + 128;                  // [16][264]   pre-LN3 output
    float*    mu   = outb + 16 * 264;            // [16]
    float*    rsd  = mu + 16;                    // [16]

    const int tid = threadIdx.x, wave = tid >> 5, lane = tid & 31;
    const int n16 = lane & 15, hf = lane >> 4;
    const int bt0 = blockIdx.x * 16;

    if (tid < 128) wf2[tid] = wff2[tid];

    // ---- gather 16 bt x 16 neighbours of h (f16 rows of 128B) into bufA
    {
        const int btl = tid >> 4, nb = tid & 15;
        const int btg = bt0 + btl;
        const int bb  = btg / TT;
        const int nbr = idxb[(size_t)btg * NH + nb];
        const _Float16* src = hh + ((size_t)bb * TCC + nbr) * MDNH;
        _Float16* dst = bufA + tid * LDX;
#pragma unroll
        for (int j = 0; j < 8; ++j)
            *(float4*)(dst + j * 8) = *(const float4*)(src + j * 8);
    }
    __syncthreads();

    // ---- cosine attention over NH=16 per (bt, head); 2 problems per wave pass
    {
#pragma unroll 1
        for (int it = 0; it < 4; ++it) {
            const int p = wave * 8 + it * 2 + hf;           // 0..63
            const int btl = p >> 2, head = p & 3;
            const int row0 = btl * 16, cb = head * 16, r = n16;
            const float scale = __expf(fminf(logit_scale[head], 4.6051702f)); // log(100)

            float qr[16]; float s2 = 0.f;
#pragma unroll
            for (int e = 0; e < 16; ++e) {
                qr[e] = (float)bufA[(row0 + r) * LDX + cb + e];
                s2 += qr[e] * qr[e];
            }
            const float inr = 1.f / (sqrtf(s2) + 1e-6f);

            float lg[16], mx = -3.0e38f;
#pragma unroll
            for (int j = 0; j < 16; ++j) {
                float dot = 0.f, nj = 0.f;
#pragma unroll
                for (int e = 0; e < 16; ++e) {
                    float qe = (float)bufA[(row0 + j) * LDX + cb + e];
                    dot += qr[e] * qe; nj += qe * qe;
                }
                float l = dot * inr / (sqrtf(nj) + 1e-6f) * scale;
                lg[j] = l; mx = fmaxf(mx, l);
            }
            float se = 0.f;
#pragma unroll
            for (int j = 0; j < 16; ++j) { lg[j] = __expf(lg[j] - mx); se += lg[j]; }
            const float ise = 1.f / se;

            float o[16];
#pragma unroll
            for (int e = 0; e < 16; ++e) o[e] = 0.f;
#pragma unroll
            for (int j = 0; j < 16; ++j) {
                float w = lg[j] * ise;
#pragma unroll
                for (int e = 0; e < 16; ++e)
                    o[e] += w * (float)bufA[(row0 + j) * LDX + cb + e];
            }
#pragma unroll
            for (int e = 0; e < 16; ++e)
                bufB[(row0 + r) * LDX + cb + e] = (_Float16)o[e];
        }
    }
    __syncthreads();

    // ---- att_out @ w_out^T (WMMA) + residual -> bufB (pre-LN2)
    {
        v8f c[4] = {};
#pragma unroll
        for (int kt = 0; kt < 2; ++kt) {
            v16h a = load_a_rowmajor(bufB + (wave * 16) * LDX + kt * 32, LDX, lane);
#pragma unroll
            for (int nt = 0; nt < 4; ++nt) {
                v16h bfr = load_pk32(pk + PK_WOUT, kt * 4 + nt, lane);
                c[nt] = __builtin_amdgcn_wmma_f32_16x16x32_f16(false, a, false, bfr,
                                                               (short)0, c[nt], false, false);
            }
        }
#pragma unroll
        for (int nt = 0; nt < 4; ++nt)
#pragma unroll
            for (int i = 0; i < 8; ++i) {
                int row = wave * 16 + i + 8 * hf, col = nt * 16 + n16;
                float v = c[nt][i] + (float)bufA[row * LDX + col];
                bufB[row * LDX + col] = (_Float16)v;
            }
    }
    __syncthreads();

    // ---- LN2 per row -> bufA
    {
        float s = 0.f, s2 = 0.f;
#pragma unroll
        for (int d = 0; d < 64; ++d) { float v = (float)bufB[tid * LDX + d]; s += v; s2 += v * v; }
        float m  = s * (1.f / 64.f);
        float rs = rsqrtf(s2 * (1.f / 64.f) - m * m + 1e-5f);
#pragma unroll
        for (int d = 0; d < 64; ++d)
            bufA[tid * LDX + d] = (_Float16)(((float)bufB[tid * LDX + d] - m) * rs * g2[d] + b2[d]);
    }
    __syncthreads();

    // ---- ff1 (WMMA, 64->128, leaky) -> yscr
    {
        v8f y[8] = {};
#pragma unroll
        for (int kt = 0; kt < 2; ++kt) {
            v16h a = load_a_rowmajor(bufA + (wave * 16) * LDX + kt * 32, LDX, lane);
#pragma unroll
            for (int nt = 0; nt < 8; ++nt) {
                v16h bfr = load_pk32(pk + PK_WFF1, kt * 8 + nt, lane);
                y[nt] = __builtin_amdgcn_wmma_f32_16x16x32_f16(false, a, false, bfr,
                                                               (short)0, y[nt], false, false);
            }
        }
#pragma unroll
        for (int nt = 0; nt < 8; ++nt)
#pragma unroll
            for (int i = 0; i < 8; ++i)
                yscr[(wave * 16 + i + 8 * hf) * LDY + nt * 16 + n16] = (_Float16)leaky(y[nt][i]);
    }
    __syncthreads();

    // ---- ff2 (128->1) + leaky + broadcast add into bufA
    {
        float acc = 0.f;
#pragma unroll
        for (int k = 0; k < 128; ++k) acc += (float)yscr[tid * LDY + k] * wf2[k];
        float s = leaky(acc);
#pragma unroll
        for (int d = 0; d < 64; ++d)
            bufA[tid * LDX + d] = (_Float16)((float)bufA[tid * LDX + d] + s);
    }
    __syncthreads();

    // ---- neighbour-mix conv: C[16,64] = conv_w[16,16(pad32)] x X[16(pad32),64]
    {
        v16h afr = load_pk32(pk + PK_CONV, 0, lane);
#pragma unroll
        for (int j = 0; j < 8; ++j) {
            const int job = wave * 8 + j;            // 16 bt x 4 n-tiles = 64 jobs
            const int btl = job >> 2, nt = job & 3;
            v16h bfr;
#pragma unroll
            for (int i = 0; i < 16; ++i) {
                int k = hf * 16 + i;                 // B lane mapping; k>=16 is the zero pad
                bfr[i] = (k < 16) ? bufA[(btl * 16 + k) * LDX + nt * 16 + n16] : (_Float16)0.f;
            }
            v8f cz = {};
            v8f c = __builtin_amdgcn_wmma_f32_16x16x32_f16(false, afr, false, bfr,
                                                           (short)0, cz, false, false);
#pragma unroll
            for (int i = 0; i < 8; ++i)
                bufB[(btl * 16 + i + 8 * hf) * LDX + nt * 16 + n16] = (_Float16)leaky(c[i]);
        }
    }
    __syncthreads();

    // ---- w_feat: [16 bt, 1024] x [1024, 256] (WMMA; B streamed from packed L2)
    {
        v8f c0 = {}, c1 = {};
        const int nt0 = wave * 2, nt1 = wave * 2 + 1;
#pragma unroll 4
        for (int kt = 0; kt < 32; ++kt) {
            // A row m = bt (lane&15); flat k = neighbour*64 + d inside bufB rows
            const int k0 = kt * 32 + hf * 8;
            const int k1 = k0 + 16;
            v16h a;
            *(float4*)&a       = *(const float4*)(bufB + (n16 * 16 + (k0 >> 6)) * LDX + (k0 & 63));
            *((float4*)&a + 1) = *(const float4*)(bufB + (n16 * 16 + (k1 >> 6)) * LDX + (k1 & 63));
            v16h b0 = load_pk32(pk + PK_WFEAT, kt * 16 + nt0, lane);
            v16h b1 = load_pk32(pk + PK_WFEAT, kt * 16 + nt1, lane);
            c0 = __builtin_amdgcn_wmma_f32_16x16x32_f16(false, a, false, b0, (short)0, c0, false, false);
            c1 = __builtin_amdgcn_wmma_f32_16x16x32_f16(false, a, false, b1, (short)0, c1, false, false);
        }
#pragma unroll
        for (int i = 0; i < 8; ++i) {
            outb[(i + 8 * hf) * 264 + nt0 * 16 + n16] = leaky(c0[i]);
            outb[(i + 8 * hf) * 264 + nt1 * 16 + n16] = leaky(c1[i]);
        }
    }
    __syncthreads();

    // ---- LN3 over MD=256 and writeout
    if (tid < 16) {
        float s = 0.f, s2 = 0.f;
        for (int d = 0; d < 256; ++d) { float v = outb[tid * 264 + d]; s += v; s2 += v * v; }
        float m = s * (1.f / 256.f);
        mu[tid]  = m;
        rsd[tid] = rsqrtf(s2 * (1.f / 256.f) - m * m + 1e-5f);
    }
    __syncthreads();
    for (int e = tid; e < 16 * 256; e += 256) {
        int row = e >> 8, col = e & 255;
        out[(size_t)(bt0 + row) * MD + col] =
            (outb[row * 264 + col] - mu[row]) * rsd[row] * g3[col] + b3[col];
    }
}

// ---------------- host launcher ----------------------------------------------
extern "C" void kernel_launch(void* const* d_in, const int* in_sizes, int n_in,
                              void* d_out, int out_size, void* d_ws, size_t ws_size,
                              hipStream_t stream) {
    (void)in_sizes; (void)n_in; (void)out_size; (void)ws_size;
    const float* coords       = (const float*)d_in[1];
    const float* x_cross      = (const float*)d_in[2];
    const float* coords_cross = (const float*)d_in[3];
    const float* w_red        = (const float*)d_in[4];
    const float* ln1_g        = (const float*)d_in[5];
    const float* ln1_b        = (const float*)d_in[6];
    const float* logit_scale  = (const float*)d_in[7];
    const float* w_out        = (const float*)d_in[8];
    const float* ln2_g        = (const float*)d_in[9];
    const float* ln2_b        = (const float*)d_in[10];
    const float* w_ff1        = (const float*)d_in[11];
    const float* w_ff2        = (const float*)d_in[12];
    const float* conv_w       = (const float*)d_in[13];
    const float* w_feat       = (const float*)d_in[14];
    const float* ln3_g        = (const float*)d_in[15];
    const float* ln3_b        = (const float*)d_in[16];
    float* out = (float*)d_out;

    char* ws = (char*)d_ws;
    _Float16* hh   = (_Float16*)(ws + WS_HH);
    int*      idxb = (int*)(ws + WS_IDX);
    _Float16* pkb  = (_Float16*)(ws + WS_PK);

    k_pack<<<PK_FRAGS, 32, 0, stream>>>(w_red, w_out, w_ff1, conv_w, w_feat, pkb);
    k_reduce_ln1<<<BTC / 64, 128, 0, stream>>>(x_cross, pkb, ln1_g, ln1_b, hh);
    k_knn<<<BT / 256, 256, 0, stream>>>(coords, coords_cross, idxb);

    const size_t smem = (size_t)(2 * 256 * LDX * sizeof(_Float16)   // bufA + bufB
                                 + 256 * LDY * sizeof(_Float16)     // yscr
                                 + 128 * sizeof(float)              // wf2
                                 + 16 * 264 * sizeof(float)         // outb
                                 + 32 * sizeof(float));             // mu + rsd
    k_fused<<<BT / 16, 256, smem, stream>>>(hh, idxb, pkb, logit_scale, ln2_g, ln2_b,
                                            w_ff2, ln3_g, ln3_b, out);
}
